// Net_90563680403530
// MI455X (gfx1250) — matrix operand order repaired and checked by reading, not compiled, not run on hardware
//
#include <hip/hip_runtime.h>
#include <hip/hip_bf16.h>

// Fused 8-layer tanh RNN + linear head for MI455X (gfx1250, wave32).
// One wave owns a 16-batch tile; all hidden state lives in VGPRs as packed f16.
//
// Per (layer, timestep): ONE v_wmma_f32_16x16x32_f16, K permuted so that the
// B operand needs no cross-lane data movement at all:
//   K0..7   = h_{l-1}[0..7]   (lo lanes, local)    A lo e0..7  = W_ih[:,0..7]
//   K8..15  = h_l(t-1)[0..7]  (lo lanes, local)    A lo e8..15 = W_hh[:,0..7]
//   K16..23 = h_{l-1}[8..15]  (hi lanes, local)    A hi e0..7  = W_ih[:,8..9],0...,bias@e7
//   K24..31 = h_l(t-1)[8..15] (hi lanes, local)    A hi e8..15 = W_hh[:,8..9],0...
// The bias rides A column K23 against h_prev[15] forced to 1.0h in hi lanes.
// Epilogue per layer: 8x v_tanh_f32 + 4x v_cvt_pk_rtz_f16_f32 + 1x v_or.

#define B_SZ 512
#define T_SZ 4096
#define H_SZ 10
#define L_SZ 8

typedef _Float16     v16h __attribute__((ext_vector_type(16)));
typedef float        v8f  __attribute__((ext_vector_type(8)));
typedef unsigned int v8u  __attribute__((ext_vector_type(8)));

__device__ __forceinline__ float tanh_fast(float x) {
#if __has_builtin(__builtin_amdgcn_tanhf)
  return __builtin_amdgcn_tanhf(x);
#else
  float e = __builtin_amdgcn_exp2f(x * 2.885390081777927f);
  return __builtin_fmaf(-2.f, __builtin_amdgcn_rcpf(e + 1.f), 1.f);
#endif
}

// lane <-> lane^16 exchange (used only for the y-head reduction and h1 output)
__device__ __forceinline__ unsigned swap16u(unsigned v) {
#if __has_builtin(__builtin_amdgcn_permlanex16)
  return __builtin_amdgcn_permlanex16(v, v, 0x76543210u, 0xfedcba98u, false, false);
#else
  return (unsigned)__builtin_amdgcn_ds_swizzle((int)v, 0x401f); // SWAPX16
#endif
}
__device__ __forceinline__ float swap16f(float v) {
  return __uint_as_float(swap16u(__float_as_uint(v)));
}

__device__ __forceinline__ unsigned pack2(float a, float b) {
#if __has_builtin(__builtin_amdgcn_cvt_pkrtz)
  auto h = __builtin_amdgcn_cvt_pkrtz(a, b);
  return __builtin_bit_cast(unsigned, h);
#else
  _Float16 h2[2] = {(_Float16)a, (_Float16)b};
  unsigned u;
  __builtin_memcpy(&u, h2, 4);
  return u;
#endif
}

__device__ __forceinline__ float lo16f(unsigned u) {
  unsigned short s = (unsigned short)(u & 0xffffu);
  _Float16 h;
  __builtin_memcpy(&h, &s, 2);
  return (float)h;
}
__device__ __forceinline__ float hi16f(unsigned u) { return lo16f(u >> 16); }

__global__ __launch_bounds__(32) void rnn_fused(
    const float* __restrict__ x, const float* __restrict__ h0,
    const float* __restrict__ w_ih0, const float* __restrict__ w_ih_rest,
    const float* __restrict__ w_hh, const float* __restrict__ b_ih,
    const float* __restrict__ b_hh, const float* __restrict__ w_lin,
    const float* __restrict__ b_lin, float* __restrict__ out) {
  const int  lane = threadIdx.x;     // 0..31
  const int  brow = lane & 15;       // A-matrix row j' / batch row in tile
  const bool lo   = lane < 16;
  const int  b    = blockIdx.x * 16 + brow;

  // ---- A operands (per-wave constants), K-permuted as described above ----
  v16h A[L_SZ];
#pragma unroll
  for (int l = 0; l < L_SZ; ++l) {
#pragma unroll
    for (int e = 0; e < 16; ++e) {
      float v = 0.f;
      if (brow < H_SZ) {
        const int jb = lo ? 0 : 8;
        if (e < 8) { // e0..7: W_ih[:, jb+e] (hi e7 = bias vs h[15]=1)
          if (jb + e < H_SZ)
            v = (l == 0) ? 0.f : w_ih_rest[((l - 1) * H_SZ + brow) * H_SZ + jb + e];
          else if (!lo && e == 7)
            v = (l == 0) ? 0.f : (b_ih[l * H_SZ + brow] + b_hh[l * H_SZ + brow]);
        } else {  // e8..15: W_hh[:, jb+e-8]
          if (jb + e - 8 < H_SZ) v = w_hh[(l * H_SZ + brow) * H_SZ + jb + e - 8];
        }
      }
      A[l][e] = (_Float16)v;
    }
  }

  // ---- layer-0 input projection constants (D_IN == 1); bias in C operand ----
  float w0[8], c0b[8];
#pragma unroll
  for (int r = 0; r < 8; ++r) {
    int j  = lo ? r : 8 + r;
    w0[r]  = (j < H_SZ) ? w_ih0[j] : 0.f;
    c0b[r] = (j < H_SZ) ? (b_ih[j] + b_hh[j]) : 0.f;
  }

  // ---- linear-head constants: this lane's half of w_lin ----
  float wlp[8];
#pragma unroll
  for (int r = 0; r < 8; ++r) {
    int j  = lo ? r : 8 + r;
    wlp[r] = (j < H_SZ) ? w_lin[j] : 0.f;
  }
  const float blin = b_lin[0];

  // ---- packed hidden state: lo lanes hold h[0..7] pairs, hi lanes h[8..15] ----
  const unsigned hibias = lo ? 0u : 0x3C000000u; // h[15] := 1.0h (bias slot K23)
  unsigned lp[L_SZ][4];
#pragma unroll
  for (int l = 0; l < L_SZ; ++l) {
    const int jb = lo ? 0 : 8;
#pragma unroll
    for (int r = 0; r < 4; ++r) {
      int   j0 = jb + 2 * r, j1 = j0 + 1;
      float a0 = (j0 < H_SZ) ? h0[((size_t)l * B_SZ + b) * H_SZ + j0] : 0.f;
      float a1 = (j1 < H_SZ) ? h0[((size_t)l * B_SZ + b) * H_SZ + j1] : 0.f;
      lp[l][r] = pack2(a0, a1);
    }
    lp[l][3] |= hibias;
  }

  const float* xrow  = x + (size_t)b * T_SZ;
  float*       ybase = out + (size_t)b * T_SZ;
  v8f          czero = {0.f, 0.f, 0.f, 0.f, 0.f, 0.f, 0.f, 0.f};
  float        xcur  = xrow[0];

  for (int t = 0; t < T_SZ; ++t) {
    const float xnext = xrow[(t + 1) & (T_SZ - 1)]; // prefetch off the chain
    float tv[8];

    // ---- layer 0: C = w_ih0*x + bias; A0 lo-half is zero so B[0..3] is x0 ----
    v8f c0;
#pragma unroll
    for (int r = 0; r < 8; ++r) c0[r] = __builtin_fmaf(w0[r], xcur, c0b[r]);
    v8u bu;
#pragma unroll
    for (int i = 0; i < 4; ++i) { bu[i] = lp[0][i]; bu[4 + i] = lp[0][i]; }
    v8f d = __builtin_amdgcn_wmma_f32_16x16x32_f16(
        false, A[0], false, __builtin_bit_cast(v16h, bu), (short)0, c0, false, false);
#pragma unroll
    for (int r = 0; r < 8; ++r) tv[r] = tanh_fast(d[r]);
    lp[0][0] = pack2(tv[0], tv[1]);
    lp[0][1] = pack2(tv[2], tv[3]);
    lp[0][2] = pack2(tv[4], tv[5]);
    lp[0][3] = pack2(tv[6], tv[7]) | hibias;

    // ---- layers 1..7: B = [lp_{l-1}(t) | lp_l(t-1)], all lane-local ----
#pragma unroll
    for (int l = 1; l < L_SZ; ++l) {
      v8u bb;
#pragma unroll
      for (int i = 0; i < 4; ++i) { bb[i] = lp[l - 1][i]; bb[4 + i] = lp[l][i]; }
      d = __builtin_amdgcn_wmma_f32_16x16x32_f16(
          false, A[l], false, __builtin_bit_cast(v16h, bb), (short)0, czero,
          false, false);
#pragma unroll
      for (int r = 0; r < 8; ++r) tv[r] = tanh_fast(d[r]);
      lp[l][0] = pack2(tv[0], tv[1]);
      lp[l][1] = pack2(tv[2], tv[3]);
      lp[l][2] = pack2(tv[4], tv[5]);
      lp[l][3] = pack2(tv[6], tv[7]) | hibias;
    }

    // ---- linear head: halves dot locally, one lane exchange combines ----
    float part = 0.f;
#pragma unroll
    for (int r = 0; r < 8; ++r) part = __builtin_fmaf(tv[r], wlp[r], part);
    float yo = part + swap16f(part) + blin;
    if (lo) ybase[t] = yo;

    xcur = xnext;
  }

  // ---- final hidden states h1[l][b][j] (reassemble halves once) ----
#pragma unroll
  for (int l = 0; l < L_SZ; ++l) {
    unsigned sp[2];
    sp[0] = swap16u(lp[l][0]); // lo lanes receive h[8..11] pairs
    sp[1] = swap16u(lp[l][1]);
    if (lo) {
      float* p = out + (size_t)B_SZ * T_SZ + ((size_t)l * B_SZ + b) * H_SZ;
      p[0] = lo16f(lp[l][0]); p[1] = hi16f(lp[l][0]);
      p[2] = lo16f(lp[l][1]); p[3] = hi16f(lp[l][1]);
      p[4] = lo16f(lp[l][2]); p[5] = hi16f(lp[l][2]);
      p[6] = lo16f(lp[l][3]); p[7] = hi16f(lp[l][3]);
      p[8] = lo16f(sp[0]);    p[9] = hi16f(sp[0]);
    }
  }
}

extern "C" void kernel_launch(void* const* d_in, const int* in_sizes, int n_in,
                              void* d_out, int out_size, void* d_ws, size_t ws_size,
                              hipStream_t stream) {
  const float* x         = (const float*)d_in[0];
  const float* h0        = (const float*)d_in[1];
  const float* w_ih0     = (const float*)d_in[2];
  const float* w_ih_rest = (const float*)d_in[3];
  const float* w_hh      = (const float*)d_in[4];
  const float* b_ih      = (const float*)d_in[5];
  const float* b_hh      = (const float*)d_in[6];
  const float* w_lin     = (const float*)d_in[7];
  const float* b_lin     = (const float*)d_in[8];
  rnn_fused<<<dim3(B_SZ / 16), dim3(32), 0, stream>>>(
      x, h0, w_ih0, w_ih_rest, w_hh, b_ih, b_hh, w_lin, b_lin, (float*)d_out);
}
